// mmRNN_51488067944780
// MI455X (gfx1250) — matrix operand order repaired and checked by reading, not compiled, not run on hardware
//
#include <hip/hip_runtime.h>
#include <hip/hip_bf16.h>

// ---------------------------------------------------------------------------
// Shapes (compile-time constants from the reference)
// ---------------------------------------------------------------------------
#define B_DIM  32
#define S_LEN  2048
#define D_DIM  512
#define H_DIM  512
#define G4_DIM 2048      // 4*H
#define O_DIM  512
#define M_ROWS (B_DIM * S_LEN)   // 65536

typedef float  v2f   __attribute__((ext_vector_type(2)));
typedef float  v8f   __attribute__((ext_vector_type(8)));
typedef __bf16 v8bf  __attribute__((ext_vector_type(8)));
typedef __bf16 v16bf __attribute__((ext_vector_type(16)));

#if __has_builtin(__builtin_amdgcn_load_to_lds)
#define ASYNC_LDS 1
#endif

__device__ __forceinline__ float sigf(float x) { return 1.0f / (1.0f + __expf(-x)); }

__device__ __forceinline__ v8f zero8() {
  v8f z;
#pragma unroll
  for (int i = 0; i < 8; ++i) z[i] = 0.0f;
  return z;
}

// 16-byte global -> LDS copy; async-to-LDS path on gfx1250 when available.
__device__ __forceinline__ void copy16_g2l(const void* g, void* l) {
#ifdef ASYNC_LDS
  __builtin_amdgcn_load_to_lds(
      (void*)(uintptr_t)g,
      (__attribute__((address_space(3))) void*)(unsigned)(uintptr_t)l,
      16, 0, 0);
#else
  *(uint4*)l = *(const uint4*)g;
#endif
}

__device__ __forceinline__ void wait_async_lds() {
#ifdef ASYNC_LDS
#if __has_builtin(__builtin_amdgcn_s_wait_asynccnt)
  __builtin_amdgcn_s_wait_asynccnt(0);
#else
  asm volatile("s_wait_asynccnt 0" ::: "memory");
#endif
#endif
}

// ---------------------------------------------------------------------------
// Kernel 0: zero the inter-workgroup barrier counters (graph-replay safe).
// ---------------------------------------------------------------------------
__global__ void zero_i32_kernel(int* p, int n) {
  int i = blockIdx.x * 256 + threadIdx.x;
  if (i < n) p[i] = 0;
}

// ---------------------------------------------------------------------------
// Kernel 1: fp32 GEMM  C[M,N] = A[M,K] @ W[K,N] + bias[N]
// V_WMMA_F32_16X16X4_F32.  Tile 64x128, BK=32, 8 waves, wave = 16x64 strip.
// Double-buffered LDS tiles; A staged via async-to-LDS, W pair-packed so
// every B fragment is one aligned ds_load_b64. NT stores for streamed output.
// ---------------------------------------------------------------------------
__global__ __launch_bounds__(256)
void gemm_f32_wmma(const float* __restrict__ A, const float* __restrict__ W,
                   const float* __restrict__ bias, float* __restrict__ C,
                   int M, int N, int K) {
  __shared__ float As[2][64][36];     // row-major, bank-skewed
  __shared__ float WsP[2][16][288];   // [k/2][n*2 + (k&1)]; 288 = 256+32

  const int tid  = threadIdx.x;
  const int wave = tid >> 5;
  const int lane = tid & 31;
  const int hi   = lane >> 4;
  const int ln   = lane & 15;
  const int mw   = wave >> 1;      // 0..3
  const int nw   = wave & 1;       // 0..1
  const int mblk = blockIdx.y * 64;
  const int nblk = blockIdx.x * 128;

  auto stage_tile = [&](int kb, int buf) {
    // A tile: 64x32 floats = 512 x 16B; 2 async copies per thread
#pragma unroll
    for (int r = 0; r < 2; ++r) {
      int q = tid + 256 * r;
      int row = q >> 3, c4 = (q & 7) * 4;
      copy16_g2l(A + (size_t)(mblk + row) * K + kb + c4, &As[buf][row][c4]);
    }
    // W tile: batch the 4 global loads, then pair-packed LDS scatter
    float4 wv[4];
#pragma unroll
    for (int r = 0; r < 4; ++r) {
      int q = tid + 256 * r;
      int k = q >> 5, c4 = (q & 31) * 4;
      wv[r] = *(const float4*)(W + (size_t)(kb + k) * N + nblk + c4);
    }
#pragma unroll
    for (int r = 0; r < 4; ++r) {
      int q = tid + 256 * r;
      int k = q >> 5, c4 = (q & 31) * 4;
      const int kp = k >> 1, c = k & 1;
      WsP[buf][kp][(c4 + 0) * 2 + c] = wv[r].x;
      WsP[buf][kp][(c4 + 1) * 2 + c] = wv[r].y;
      WsP[buf][kp][(c4 + 2) * 2 + c] = wv[r].z;
      WsP[buf][kp][(c4 + 3) * 2 + c] = wv[r].w;
    }
  };

  v8f acc[4];
#pragma unroll
  for (int j = 0; j < 4; ++j) acc[j] = zero8();

  stage_tile(0, 0);
  const int ntiles = K >> 5;
  for (int t = 0; t < ntiles; ++t) {
    const int buf = t & 1;
    wait_async_lds();        // tile t's async copies complete
    __syncthreads();         // tile t visible to all waves
    if (t + 1 < ntiles) stage_tile((t + 1) << 5, buf ^ 1);  // overlap with compute

#pragma unroll
    for (int kk = 0; kk < 32; kk += 4) {
      // f32 A frag 16x4: lanes0-15 K=kk,kk+1 ; lanes16-31 K=kk+2,kk+3
      const v2f a = *(const v2f*)&As[buf][mw * 16 + ln][kk + hi * 2];
#pragma unroll
      for (int j = 0; j < 4; ++j) {
        const v2f b =
            *(const v2f*)&WsP[buf][(kk >> 1) + hi][(nw * 64 + j * 16 + ln) * 2];
        acc[j] = __builtin_amdgcn_wmma_f32_16x16x4_f32(
            false, a, false, b, (short)0, acc[j], false, false);
      }
    }
  }

#pragma unroll
  for (int j = 0; j < 4; ++j)
#pragma unroll
    for (int i = 0; i < 8; ++i) {
      int row = mblk + mw * 16 + hi * 8 + i;
      int col = nblk + nw * 64 + j * 16 + ln;
      __builtin_nontemporal_store(acc[j][i] + bias[col], &C[(size_t)row * N + col]);
    }
}

// ---------------------------------------------------------------------------
// Inter-workgroup barrier among the 16 column-slice WGs of a batch group.
// ---------------------------------------------------------------------------
__device__ __forceinline__ void group_barrier(int* bars, int g, int s, int p, int tid) {
  __syncthreads();
  if (tid == 0) {
    __threadfence();
    int* ctr = bars + g * (3 * S_LEN) + s * 3 + p;
    __hip_atomic_fetch_add(ctr, 1, __ATOMIC_ACQ_REL, __HIP_MEMORY_SCOPE_AGENT);
    while (__hip_atomic_load(ctr, __ATOMIC_ACQUIRE, __HIP_MEMORY_SCOPE_AGENT) < 16) {
      __builtin_amdgcn_s_sleep(1);
    }
  }
  __syncthreads();
}

// bf16 16x16x32 fragment matmul. A = hstg[16][512] bf16 (two 16B loads +
// shuffle per frag); B from fragment-major packed weights (one 32B load).
template <int NKB>
__device__ __forceinline__ v8f mm_bf16_frag(const __bf16* __restrict__ hstg,
                                            const __bf16* __restrict__ Wp,
                                            int kb0, int lane, v8f acc) {
  const int hi = lane >> 4, ln = lane & 15;
#pragma unroll
  for (int kbi = 0; kbi < NKB; ++kbi) {
    const int kb = (kb0 + kbi) * 32;
    const v8bf a0 = *(const v8bf*)&hstg[ln * H_DIM + kb + hi * 8];
    const v8bf a1 = *(const v8bf*)&hstg[ln * H_DIM + kb + 16 + hi * 8];
    const v16bf a = __builtin_shufflevector(a0, a1,
        0, 1, 2, 3, 4, 5, 6, 7, 8, 9, 10, 11, 12, 13, 14, 15);
    const v16bf b = *(const v16bf*)&Wp[((size_t)(kb0 + kbi) * 32 + lane) * 16];
    acc = __builtin_amdgcn_wmma_f32_16x16x32_bf16(
        false, a, false, b, (short)0, acc, false, false);
  }
  return acc;
}

// ---------------------------------------------------------------------------
// Kernel 2: serial recurrence. 32 WGs = 2 batch groups x 16 column slices.
// Wh slices live in LDS (bf16, fragment-major) for all 2048 steps.
// h exchange between WGs is bf16 through global, re-staged with async LDS.
// ---------------------------------------------------------------------------
__global__ __launch_bounds__(256)
void recurrence_kernel(const float* __restrict__ XZL, const float* __restrict__ XZO,
                       const float* __restrict__ TS,
                       const float* __restrict__ Whl, const float* __restrict__ Who,
                       float* __restrict__ HS,
                       __bf16* __restrict__ HcurB, __bf16* __restrict__ HmemB,
                       __bf16* __restrict__ TmpB,
                       int* __restrict__ bars,
                       float* __restrict__ hlast, float* __restrict__ clast) {
  extern __shared__ char smem[];
  __bf16* WlP  = (__bf16*)smem;             // [8 ntile][16 kb][32 lane][16]  128KB
  __bf16* WoP  = WlP + 8 * 16 * 32 * 16;    // [2 ntile][16 kb][32 lane][16]   32KB
  __bf16* hstg = WoP + 2 * 16 * 32 * 16;    // [16][512]                       16KB
  float*  pbuf = (float*)(hstg + 16 * 512); // [8 tile][16][16] z / partials    8KB
  float*  cbuf = pbuf + 8 * 16 * 16;        // [16][32] c state (fp32)
  float*  hmb  = cbuf + 16 * 32;            // [16][32] h_mem (own cols, fp32)
  float*  k1b  = hmb + 16 * 32;             // [16][32] k1 (own cols, fp32)
  float*  tsb  = k1b + 16 * 32;             // [16]

  const int tid  = threadIdx.x;
  const int wave = tid >> 5;
  const int lane = tid & 31;
  const int hi   = lane >> 4;
  const int ln   = lane & 15;
  const int g    = blockIdx.x >> 4;         // batch group 0..1
  const int w    = blockIdx.x & 15;         // column slice 0..15
  const int col0 = w * 32;

  // --- one-time: pack weight slices into fragment-major bf16 LDS images ---
  // B frag element e of lane: K = kb*32 + (e&7) + (lane>>4)*8 + (e>>3)*16, N = ntile*16 + (lane&15)
  for (int idx = tid; idx < 8 * 16 * 32 * 16; idx += 256) {
    int e = idx & 15, lp = (idx >> 4) & 31, kbi = (idx >> 9) & 15, t = idx >> 13;
    int k  = kbi * 32 + (e & 7) + (lp >> 4) * 8 + (e >> 3) * 16;
    int lc = t * 16 + (lp & 15);            // local col 0..127
    int gate = lc >> 5, j = lc & 31;
    WlP[idx] = (__bf16)Whl[(size_t)k * G4_DIM + gate * H_DIM + col0 + j];
  }
  for (int idx = tid; idx < 2 * 16 * 32 * 16; idx += 256) {
    int e = idx & 15, lp = (idx >> 4) & 31, kbi = (idx >> 9) & 15, t = idx >> 13;
    int k = kbi * 32 + (e & 7) + (lp >> 4) * 8 + (e >> 3) * 16;
    int col = col0 + t * 16 + (lp & 15);
    WoP[idx] = (__bf16)Who[(size_t)k * H_DIM + col];
  }
  for (int idx = tid; idx < 16 * 512; idx += 256) hstg[idx] = (__bf16)0.0f; // h0=0
  for (int idx = tid; idx < 16 * 32;  idx += 256) cbuf[idx] = 0.0f;         // c0=0
  __syncthreads();

  for (int s = 0; s < S_LEN; ++s) {
    if (tid < 16) tsb[tid] = TS[(size_t)(g * 16 + tid) * S_LEN + s];
    __syncthreads();

    // ---- Stage A: z = xz_lstm + h @ Whl_slice (8 waves, N=16 each, K=512) ----
    {
      v8f acc = mm_bf16_frag<16>(hstg, WlP + wave * (16 * 32 * 16), 0, lane, zero8());
      const int gate = wave >> 1, half = wave & 1;
      const int colg = gate * H_DIM + col0 + half * 16 + ln;
#pragma unroll
      for (int i = 0; i < 8; ++i) {
        int bl = hi * 8 + i;
        int bg = g * 16 + bl;
        pbuf[(wave * 16 + bl) * 16 + ln] =
            acc[i] + XZL[((size_t)bg * S_LEN + s) * G4_DIM + colg];
        if (s + 1 < S_LEN)
          __builtin_prefetch(&XZL[((size_t)bg * S_LEN + s + 1) * G4_DIM + colg], 0, 1);
      }
    }
    __syncthreads();

    // ---- elementwise LSTM cell; publish h_mem slice (bf16) ----
    for (int e = tid; e < 512; e += 256) {
      int b = e >> 5, j = e & 31, half = j >> 4, jn = j & 15;
      float iv = pbuf[((0 + half) * 16 + b) * 16 + jn];
      float fv = pbuf[((2 + half) * 16 + b) * 16 + jn];
      float gv = pbuf[((4 + half) * 16 + b) * 16 + jn];
      float ov = pbuf[((6 + half) * 16 + b) * 16 + jn];
      float c_new = sigf(fv) * cbuf[e] + sigf(iv) * tanhf(gv);
      float h_mem = sigf(ov) * tanhf(c_new);
      cbuf[e] = c_new;
      hmb[e]  = h_mem;
      HmemB[(g * 16 + b) * H_DIM + col0 + j] = (__bf16)h_mem;
    }
    group_barrier(bars, g, s, 0, tid);

    // gather full h_mem (16KB) -> LDS via async copies
    {
      const __bf16* src = HmemB + (size_t)g * 16 * H_DIM;
      for (int idx = tid; idx < (16 * H_DIM) / 8; idx += 256)
        copy16_g2l(src + idx * 8, hstg + idx * 8);
      wait_async_lds();
    }
    __syncthreads();

    // ---- Stage B: k1 partials (8 waves: 2 N-tiles x 4 K-chunks) ----
    {
      const int nt = wave & 1, kc = wave >> 1;
      v8f acc = mm_bf16_frag<4>(hstg, WoP + nt * (16 * 32 * 16), kc * 4, lane, zero8());
#pragma unroll
      for (int i = 0; i < 8; ++i)
        pbuf[(wave * 16 + hi * 8 + i) * 16 + ln] = acc[i];
    }
    __syncthreads();
    for (int e = tid; e < 512; e += 256) {
      int b = e >> 5, j = e & 31, half = j >> 4, jn = j & 15;
      float sum = pbuf[((0 + half) * 16 + b) * 16 + jn]
                + pbuf[((2 + half) * 16 + b) * 16 + jn]
                + pbuf[((4 + half) * 16 + b) * 16 + jn]
                + pbuf[((6 + half) * 16 + b) * 16 + jn];
      float zo = XZO[((size_t)(g * 16 + b) * S_LEN + s) * H_DIM + col0 + j];
      float k1 = tanhf(zo + sum);
      k1b[e] = k1;
      TmpB[(g * 16 + b) * H_DIM + col0 + j] = (__bf16)(hmb[e] + tsb[b] * k1);
    }
    group_barrier(bars, g, s, 1, tid);

    {
      const __bf16* src = TmpB + (size_t)g * 16 * H_DIM;
      for (int idx = tid; idx < (16 * H_DIM) / 8; idx += 256)
        copy16_g2l(src + idx * 8, hstg + idx * 8);
      wait_async_lds();
    }
    __syncthreads();

    // ---- Stage C: k2 partials + Heun update ----
    {
      const int nt = wave & 1, kc = wave >> 1;
      v8f acc = mm_bf16_frag<4>(hstg, WoP + nt * (16 * 32 * 16), kc * 4, lane, zero8());
#pragma unroll
      for (int i = 0; i < 8; ++i)
        pbuf[(wave * 16 + hi * 8 + i) * 16 + ln] = acc[i];
    }
    __syncthreads();
    for (int e = tid; e < 512; e += 256) {
      int b = e >> 5, j = e & 31, half = j >> 4, jn = j & 15;
      float sum = pbuf[((0 + half) * 16 + b) * 16 + jn]
                + pbuf[((2 + half) * 16 + b) * 16 + jn]
                + pbuf[((4 + half) * 16 + b) * 16 + jn]
                + pbuf[((6 + half) * 16 + b) * 16 + jn];
      int bg = g * 16 + b;
      float zo = XZO[((size_t)bg * S_LEN + s) * H_DIM + col0 + j];
      float k2 = tanhf(zo + sum);
      float hn = hmb[e] + 0.5f * tsb[b] * (k1b[e] + k2);
      HcurB[bg * H_DIM + col0 + j] = (__bf16)hn;
      HS[((size_t)bg * S_LEN + s) * H_DIM + col0 + j] = hn;
      if (s == S_LEN - 1) hlast[bg * H_DIM + col0 + j] = hn;
    }
    group_barrier(bars, g, s, 2, tid);

    {
      const __bf16* src = HcurB + (size_t)g * 16 * H_DIM;
      for (int idx = tid; idx < (16 * H_DIM) / 8; idx += 256)
        copy16_g2l(src + idx * 8, hstg + idx * 8);
      wait_async_lds();
    }
    __syncthreads();
  }

  for (int e = tid; e < 512; e += 256) {
    int b = e >> 5, j = e & 31;
    clast[(g * 16 + b) * H_DIM + col0 + j] = cbuf[e];
  }
}

// ---------------------------------------------------------------------------
// Host launcher
// ---------------------------------------------------------------------------
extern "C" void kernel_launch(void* const* d_in, const int* in_sizes, int n_in,
                              void* d_out, int out_size, void* d_ws, size_t ws_size,
                              hipStream_t stream) {
  (void)in_sizes; (void)n_in; (void)out_size; (void)ws_size;

  const float* x    = (const float*)d_in[0];
  const float* ts   = (const float*)d_in[1];
  const float* Wxl  = (const float*)d_in[2];
  const float* Whl  = (const float*)d_in[3];
  const float* bl   = (const float*)d_in[4];
  const float* Wxo  = (const float*)d_in[5];
  const float* Who  = (const float*)d_in[6];
  const float* bo   = (const float*)d_in[7];
  const float* Wfc  = (const float*)d_in[8];
  const float* bfc  = (const float*)d_in[9];
  float* out = (float*)d_out;

  float* XZL = (float*)d_ws;                         // [B*S, 4H] fp32
  float* XZO = XZL + (size_t)M_ROWS * G4_DIM;        // [B*S, H] fp32
  float* HS  = XZO + (size_t)M_ROWS * H_DIM;         // [B*S, H] fp32 (b,s,h)
  __bf16* HcurB = (__bf16*)(HS + (size_t)M_ROWS * H_DIM);  // [B, H] bf16
  __bf16* HmemB = HcurB + B_DIM * H_DIM;
  __bf16* TmpB  = HmemB + B_DIM * H_DIM;
  int*    bars  = (int*)(TmpB + B_DIM * H_DIM);      // [2][S][3]
  const int nbars = 2 * 3 * S_LEN;

  float* hlast = out + (size_t)M_ROWS * O_DIM;
  float* clast = hlast + B_DIM * H_DIM;

  dim3 blk(256);

  zero_i32_kernel<<<dim3((nbars + 255) / 256), blk, 0, stream>>>(bars, nbars);

  gemm_f32_wmma<<<dim3(G4_DIM / 128, M_ROWS / 64), blk, 0, stream>>>(
      x, Wxl, bl, XZL, M_ROWS, G4_DIM, D_DIM);
  gemm_f32_wmma<<<dim3(H_DIM / 128, M_ROWS / 64), blk, 0, stream>>>(
      x, Wxo, bo, XZO, M_ROWS, H_DIM, D_DIM);

  const size_t lds_bytes =
      (size_t)(8 * 16 * 32 * 16 + 2 * 16 * 32 * 16 + 16 * 512) * sizeof(__bf16) +
      (size_t)(8 * 16 * 16 + 3 * 16 * 32 + 16) * sizeof(float);
  recurrence_kernel<<<dim3(32), blk, lds_bytes, stream>>>(
      XZL, XZO, ts, Whl, Who, HS, HcurB, HmemB, TmpB, bars, hlast, clast);

  gemm_f32_wmma<<<dim3(O_DIM / 128, M_ROWS / 64), blk, 0, stream>>>(
      HS, Wfc, bfc, out, M_ROWS, O_DIM, H_DIM);
}
